// ProxyNCALoss_88519275970911
// MI455X (gfx1250) — matrix operand order repaired and checked by reading, not compiled, not run on hardware
//
#include <hip/hip_runtime.h>
#include <hip/hip_bf16.h>
#include <math.h>

typedef __attribute__((ext_vector_type(8)))  __bf16 v8bf;
typedef __attribute__((ext_vector_type(16))) __bf16 v16bf;
typedef __attribute__((ext_vector_type(8)))  float  v8f;

union BFrag {
    v16bf  v;
    v8bf   h[2];
    __bf16 e[16];
};
union BFrag8 {
    v8bf   v;
    __bf16 e[8];
};

// ---------------- Kernel 1: split xs (f32) into bf16 hi + lo ----------------
__global__ void split_bf16_kernel(const float* __restrict__ x,
                                  __bf16* __restrict__ xh,
                                  __bf16* __restrict__ xl, int n) {
    int i = blockIdx.x * blockDim.x + threadIdx.x;
    if (i < n) {
        float v  = x[i];
        __bf16 h = (__bf16)v;            // RNE
        xh[i] = h;
        xl[i] = (__bf16)(v - (float)h);  // residual
    }
}

// ---------------- Kernel 2: exact-f32 positive term g_pos[b] ----------------
__global__ void pos_kernel(const float* __restrict__ xs,
                           const int* __restrict__ ys,
                           const float* __restrict__ proxies,
                           float* __restrict__ g_pos, int B, int C) {
    int wid  = threadIdx.x >> 5;
    int lane = threadIdx.x & 31;
    int b = blockIdx.x * (blockDim.x >> 5) + wid;
    if (b >= B) return;
    int y = ys[b];
    y = (y < 0) ? 0 : ((y >= C) ? C - 1 : y);
    float4 xv = *(const float4*)(xs + (size_t)b * 128 + lane * 4);
    float4 pv = *(const float4*)(proxies + (size_t)y * 128 + lane * 4);
    float dot = xv.x * pv.x + xv.y * pv.y + xv.z * pv.z + xv.w * pv.w;
    float p2  = pv.x * pv.x + pv.y * pv.y + pv.z * pv.z + pv.w * pv.w;
    #pragma unroll
    for (int m = 1; m < 32; m <<= 1) {
        dot += __shfl_xor(dot, m, 32);
        p2  += __shfl_xor(p2,  m, 32);
    }
    if (lane == 0) g_pos[b] = dot / fmaxf(sqrtf(p2), 1e-12f);
}

// -------- Kernel 3: row-stationary bf16x3 WMMA GEMM + fused exp/row-sum -----
// Grid: (column stripes) x (row groups of 256). Wave holds A (32 rows of xs,
// hi+lo) in registers; 16 proxy columns/tile staged through double-buffered
// LDS in pre-swizzled B-fragment layout, shared by all 8 waves.
__global__ void __launch_bounds__(256)
proxy_gemm_kernel(const __bf16* __restrict__ xh, const __bf16* __restrict__ xl,
                  const float* __restrict__ proxies,
                  float* __restrict__ S_part,
                  int B, int C, int ntiles, int tpw) {
    // frag[buf][hi/lo][kk][lane][i] : lane reads its own contiguous 32B.
    __shared__ __align__(32) __bf16 frag[2][2][4][32][16];   // 16 KB
    __shared__ float norms[2][16];

    const int tid  = threadIdx.x;
    const int wid  = tid >> 5;
    const int lane = tid & 31;
    const int half = lane >> 4;
    const int lcol = lane & 15;

    const int rowbase = blockIdx.y * 256 + wid * 32;   // this wave's 32 rows
    const int tile0   = blockIdx.x * tpw;
    const int T       = min(tpw, ntiles - tile0);

    // ---- stationary A fragments: 2 row blocks x 4 K-chunks x {hi,lo} ----
    // A layout (16x32 bf16): lane = row M; elems 0..7 = K k0..k0+7,
    // elems 8..15 = K k0+16..k0+23, k0 = kk*32 + half*8.
    BFrag ah[2][4], al[2][4];
    #pragma unroll
    for (int blk = 0; blk < 2; ++blk) {
        const int row = rowbase + blk * 16 + lcol;
        const __bf16* hp = xh + (long)row * 128;
        const __bf16* lp = xl + (long)row * 128;
        #pragma unroll
        for (int kk = 0; kk < 4; ++kk) {
            const int k0 = kk * 32 + half * 8;
            ah[blk][kk].h[0] = *(const v8bf*)(hp + k0);
            ah[blk][kk].h[1] = *(const v8bf*)(hp + k0 + 16);
            al[blk][kk].h[0] = *(const v8bf*)(lp + k0);
            al[blk][kk].h[1] = *(const v8bf*)(lp + k0 + 16);
        }
    }

    // staging role of this thread: col s_col, 8 K-values starting at s_k8
    const int s_col  = tid >> 4;
    const int s_k8   = (tid & 15) * 8;
    const int s_kk   = s_k8 >> 5;
    const int s_lane = s_col + (((s_k8 >> 3) & 1) << 4);
    const int s_i0   = (s_k8 >> 1) & 8;

    float esum[2][8];
    #pragma unroll
    for (int blk = 0; blk < 2; ++blk)
        #pragma unroll
        for (int j = 0; j < 8; ++j) esum[blk][j] = 0.0f;

    auto stageLoad = [&](int tt, float4& a, float4& b) {
        const int c = tt * 16 + s_col;
        const long crow = (c < C) ? c : (C - 1);
        const float* p = proxies + crow * 128 + s_k8;
        a = *(const float4*)(p);
        b = *(const float4*)(p + 4);
    };
    auto stageStore = [&](int buf, float4 a, float4 b) {
        float vv[8] = {a.x, a.y, a.z, a.w, b.x, b.y, b.z, b.w};
        BFrag8 hb, lb;
        float p2 = 0.0f;
        #pragma unroll
        for (int q = 0; q < 8; ++q) {
            float v = vv[q];
            p2 += v * v;
            __bf16 h = (__bf16)v;
            hb.e[q] = h;
            lb.e[q] = (__bf16)(v - (float)h);
        }
        *(v8bf*)&frag[buf][0][s_kk][s_lane][s_i0] = hb.v;
        *(v8bf*)&frag[buf][1][s_kk][s_lane][s_i0] = lb.v;
        p2 += __shfl_xor(p2, 1, 32);
        p2 += __shfl_xor(p2, 2, 32);
        p2 += __shfl_xor(p2, 4, 32);
        p2 += __shfl_xor(p2, 8, 32);
        if ((tid & 15) == 0)
            norms[buf][s_col] = 1.0f / fmaxf(sqrtf(p2), 1e-12f);
    };
    auto consume = [&](int buf, int tt) {
        const float vm   = (tt * 16 + lcol < C) ? 1.0f : 0.0f;
        const float invn = norms[buf][lcol];
        v8f acc0 = {}, acc1 = {};
        v16bf bh0 = *(const v16bf*)&frag[buf][0][0][lane][0];
        v16bf bl0 = *(const v16bf*)&frag[buf][1][0][lane][0];
        #pragma unroll
        for (int kk = 0; kk < 4; ++kk) {
            v16bf bh1 = bh0, bl1 = bl0;
            if (kk < 3) {   // prefetch next chunk's fragments
                bh1 = *(const v16bf*)&frag[buf][0][kk + 1][lane][0];
                bl1 = *(const v16bf*)&frag[buf][1][kk + 1][lane][0];
            }
            // bf16x3: xh*ph + xh*pl + xl*ph, two independent acc chains
            acc0 = __builtin_amdgcn_wmma_f32_16x16x32_bf16(
                       false, ah[0][kk].v, false, bh0, (short)0, acc0, false, false);
            acc1 = __builtin_amdgcn_wmma_f32_16x16x32_bf16(
                       false, ah[1][kk].v, false, bh0, (short)0, acc1, false, false);
            acc0 = __builtin_amdgcn_wmma_f32_16x16x32_bf16(
                       false, ah[0][kk].v, false, bl0, (short)0, acc0, false, false);
            acc1 = __builtin_amdgcn_wmma_f32_16x16x32_bf16(
                       false, ah[1][kk].v, false, bl0, (short)0, acc1, false, false);
            acc0 = __builtin_amdgcn_wmma_f32_16x16x32_bf16(
                       false, al[0][kk].v, false, bh0, (short)0, acc0, false, false);
            acc1 = __builtin_amdgcn_wmma_f32_16x16x32_bf16(
                       false, al[1][kk].v, false, bh0, (short)0, acc1, false, false);
            bh0 = bh1; bl0 = bl1;
        }
        // C/D layout: VGPR j, lane -> row = blk*16 + j + 8*half, col = lcol
        #pragma unroll
        for (int j = 0; j < 8; ++j) {
            esum[0][j] += vm * __expf(2.0f * (acc0[j] * invn));
            esum[1][j] += vm * __expf(2.0f * (acc1[j] * invn));
        }
    };

    // ---- software-pipelined main loop over column tiles ----
    float4 sA, sB;
    stageLoad(tile0, sA, sB);
    stageStore(0, sA, sB);
    __syncthreads();
    for (int t = 0; t < T; ++t) {
        if (t + 1 < T) stageLoad(tile0 + t + 1, sA, sB);  // hide latency
        consume(t & 1, tile0 + t);
        if (t + 1 < T) stageStore((t + 1) & 1, sA, sB);
        __syncthreads();
    }

    // ---- one butterfly per workgroup: sum e over the 16 column lanes ----
    #pragma unroll
    for (int blk = 0; blk < 2; ++blk)
        #pragma unroll
        for (int j = 0; j < 8; ++j) {
            float e = esum[blk][j];
            e += __shfl_xor(e, 1, 32);
            e += __shfl_xor(e, 2, 32);
            e += __shfl_xor(e, 4, 32);
            e += __shfl_xor(e, 8, 32);
            esum[blk][j] = e;
        }
    if (lcol == 0) {  // lanes 0 and 16 own disjoint row halves
        float* out = S_part + (size_t)blockIdx.x * B + rowbase;
        #pragma unroll
        for (int blk = 0; blk < 2; ++blk)
            #pragma unroll
            for (int j = 0; j < 8; ++j)
                out[blk * 16 + j + (half << 3)] = esum[blk][j];
    }
}

// ------------- Kernel 4: reduce partials, per-row loss, mean ----------------
__global__ void finalize_kernel(const float* __restrict__ S_part,
                                const float* __restrict__ g_pos,
                                float* __restrict__ out, int B, int nstripe) {
    __shared__ float red[1024];
    const int b = threadIdx.x;  // blockDim.x == B
    float s = 0.0f;
    for (int t = 0; t < nstripe; ++t) s += S_part[(size_t)t * B + b];
    float gp = g_pos[b];
    float lb = logf(s - __expf(2.0f * gp)) - 2.0f * gp;
    red[b] = lb;
    __syncthreads();
    for (int st = blockDim.x >> 1; st > 0; st >>= 1) {
        if (b < st) red[b] += red[b + st];
        __syncthreads();
    }
    if (b == 0) out[0] = red[0] / (float)B;
}

extern "C" void kernel_launch(void* const* d_in, const int* in_sizes, int n_in,
                              void* d_out, int out_size, void* d_ws, size_t ws_size,
                              hipStream_t stream) {
    (void)n_in; (void)out_size; (void)ws_size;
    const float* xs      = (const float*)d_in[0];
    const int*   ys      = (const int*)d_in[1];
    const float* proxies = (const float*)d_in[2];

    const int B = in_sizes[1];            // 1024
    const int D = in_sizes[0] / B;        // 128 (layout assumes 128)
    const int C = in_sizes[2] / D;        // 100000

    char* ws = (char*)d_ws;
    const size_t nxd = (size_t)B * D;
    __bf16* xh = (__bf16*)ws;  ws += ((nxd * 2 + 255) / 256) * 256;
    __bf16* xl = (__bf16*)ws;  ws += ((nxd * 2 + 255) / 256) * 256;
    float* g_pos = (float*)ws; ws += (((size_t)B * 4 + 255) / 256) * 256;

    const int ntile16 = (C + 15) / 16;          // 6250
    const int tpw     = 16;                     // column tiles per workgroup
    const int nstripe = (ntile16 + tpw - 1) / tpw;  // 391
    float* S_part = (float*)ws;                 // nstripe * B floats (~1.6 MB)

    split_bf16_kernel<<<(int)((nxd + 255) / 256), 256, 0, stream>>>(xs, xh, xl, (int)nxd);
    pos_kernel<<<(B + 7) / 8, 256, 0, stream>>>(xs, ys, proxies, g_pos, B, C);
    dim3 ggrid(nstripe, B / 256);
    proxy_gemm_kernel<<<ggrid, 256, 0, stream>>>(xh, xl, proxies, S_part,
                                                 B, C, ntile16, tpw);
    finalize_kernel<<<1, B, 0, stream>>>(S_part, g_pos, (float*)d_out, B, nstripe);
}
